// HybridTRM_41480794145075
// MI455X (gfx1250) — matrix operand order repaired and checked by reading, not compiled, not run on hardware
//
#include <hip/hip_runtime.h>
#include <hip/hip_bf16.h>
#include <cstdint>

typedef __attribute__((ext_vector_type(16))) _Float16 v16h;
typedef __attribute__((ext_vector_type(8)))  float    v8f;
typedef unsigned int v4u __attribute__((ext_vector_type(4)));
typedef int          v8i __attribute__((ext_vector_type(8)));
typedef int          v4i __attribute__((ext_vector_type(4)));

#define LN_EPS 1e-5f

// ---------------- packed-weight workspace layout (halves) ----------------
enum : size_t {
  OFF_MPW  = 0,       // 200x32 -> KC=7 NT=2 : 7168
  OFF_SAWK = 7168,    // 32x32  -> KC=1 NT=2 : 1024
  OFF_SAWV = 8192,
  OFF_SAWO = 9216,
  OFF_CAWV = 10240,
  OFF_CAWO = 11264,
  OFF_FF1  = 12288,   // 32x64  -> KC=1 NT=4 : 2048
  OFF_FF2  = 14336,   // 64x32  -> KC=2 NT=2 : 2048
  OFF_POOL = 16384,   // 32x64  -> KC=1 NT=4 : 2048
  OFF_ZW1  = 18432,   // 192x100-> KC=6 NT=7 : 21504
  OFF_ZW2  = 39936,   // 100x64 -> KC=4 NT=4 : 8192
  OFF_YW1  = 48128,   // 128x100-> KC=4 NT=7 : 14336
  OFF_YW2  = 62464,   // 100x64 -> KC=4 NT=4 : 8192
  WPK_TOTAL = 70656   // halves (141312 bytes in d_ws)
};

// per-wave LDS stride (bytes): tok 22528 | kv 45056 | scr 8192 | ctxo 2048 | xstg 2048
#define WAVE_LDS 79872

// ---------------- helpers ----------------
__device__ __forceinline__ float gelu_f(float x){
  return 0.5f * x * (1.0f + erff(x * 0.70710678118654752f));
}

__device__ __forceinline__ void lds_fence(){
  asm volatile("s_wait_dscnt 0" ::: "memory");
}

__device__ __forceinline__ void tensor_wait(){
#if __has_builtin(__builtin_amdgcn_s_wait_tensorcnt)
  __builtin_amdgcn_s_wait_tensorcnt(0);
#else
  asm volatile("s_wait_tensorcnt 0" ::: "memory");
#endif
}

// TDM: load a rows x tile_cols f32 tile (row stride row_stride_elems) into LDS.
// OOB columns/rows (beyond rem_cols/rows) are zero-filled by the TDM per ISA.
__device__ __forceinline__ void tdm_load_2d_f32(const void* gaddr, uint32_t lds_off,
                                                uint32_t rem_cols, uint32_t rows,
                                                uint32_t tile_cols, uint32_t row_stride_elems){
#if __has_builtin(__builtin_amdgcn_tensor_load_to_lds)
  uint64_t ga = (uint64_t)(uintptr_t)gaddr;
  v4u g0;
  g0[0] = 1u;                                                  // count=1 (user D#)
  g0[1] = lds_off;                                             // lds_addr (bytes)
  g0[2] = (uint32_t)ga;                                        // global_addr[31:0]
  g0[3] = (uint32_t)((ga >> 32) & 0x1FFFFFFu) | (2u << 30);    // addr[56:32] | type=2
  v8i g1;
  g1[0] = (int)(2u << 16);                                     // data_size = 4 bytes
  g1[1] = (int)(rem_cols << 16);                               // tensor_dim0[15:0] @ bit48
  g1[2] = (int)((rem_cols >> 16) | (rows << 16));              // dim0[31:16] | dim1[15:0]
  g1[3] = (int)((rows >> 16) | (tile_cols << 16));             // dim1[31:16] | tile_dim0
  g1[4] = (int)(rows & 0xFFFFu);                               // tile_dim1 (tile_dim2=0)
  g1[5] = (int)row_stride_elems;                               // tensor_dim0_stride[31:0]
  g1[6] = 0;                                                   // stride0 hi | stride1 lo
  g1[7] = 0;
  v4i gz; gz[0] = gz[1] = gz[2] = gz[3] = 0;                   // groups 2/3 unused (2D)
  #if defined(__clang_major__) && (__clang_major__ >= 23)
  v8i gz8;
  #pragma unroll
  for (int i = 0; i < 8; ++i) gz8[i] = 0;
  __builtin_amdgcn_tensor_load_to_lds(g0, g1, gz, gz, gz8, 0);
  #else
  __builtin_amdgcn_tensor_load_to_lds(g0, g1, gz, gz, 0);
  #endif
#endif
}

__device__ __forceinline__ v8f vzero(){
  v8f z;
  #pragma unroll
  for (int r = 0; r < 8; ++r) z[r] = 0.f;
  return z;
}

__device__ __forceinline__ v8f wmma_f16(v16h a, v16h b, v8f c){
  // D = A(16x32 f16) * B(32x16 f16) + C(16x16 f32)
  return __builtin_amdgcn_wmma_f32_16x16x32_f16(false, a, false, b, (short)0, c, false, false);
}

// A fragment (16x32 f16) from f16 LDS tile, row stride ld halves, k-chunk kc.
// Layout per ISA: lanes 0-15 row=lane hold K {0..7,16..23}; lanes 16-31 hold K {8..15,24..31}.
__device__ __forceinline__ v16h ldA_h(const _Float16* base, int ld, int kc, int lane){
  int row = lane & 15;
  int kb  = kc * 32 + ((lane & 16) ? 8 : 0);
  const _Float16* p0 = base + row * ld + kb;
  v16h a;
  #pragma unroll
  for (int j = 0; j < 8; ++j) a[j] = p0[j];
  #pragma unroll
  for (int j = 0; j < 8; ++j) a[8 + j] = p0[16 + j];
  return a;
}

// Same, sourcing f32 LDS and converting to f16.
__device__ __forceinline__ v16h ldA_f(const float* base, int ld, int kc, int lane){
  int row = lane & 15;
  int kb  = kc * 32 + ((lane & 16) ? 8 : 0);
  const float* p0 = base + row * ld + kb;
  v16h a;
  #pragma unroll
  for (int j = 0; j < 8; ++j) a[j] = (_Float16)p0[j];
  #pragma unroll
  for (int j = 0; j < 8; ++j) a[8 + j] = (_Float16)p0[16 + j];
  return a;
}

// B fragment: pre-packed lane-contiguous (32B per lane) in global memory.
__device__ __forceinline__ v16h ldB(const _Float16* wp, int frag, int lane){
  return *(const v16h*)(wp + (size_t)frag * 512 + (size_t)lane * 16);
}

// Store D frag (+bias, optional gelu, column clamp) into f32 LDS tile.
// C/D layout: VGPR r -> (M = r or 8+r), lane&15 = N.
__device__ __forceinline__ void stD_f(float* base, int ld, int col0, int nt, v8f acc,
                                      const float* bias, int nvalid, bool g, int lane){
  int n  = lane & 15;
  int mb = (lane & 16) ? 8 : 0;
  int oc = nt * 16 + n;
  bool ok = (oc < nvalid);
  float bv = ok ? bias[oc] : 0.f;
  #pragma unroll
  for (int r = 0; r < 8; ++r){
    float v = ok ? (acc[r] + bv) : 0.f;
    if (g) v = gelu_f(v);
    base[(mb + r) * ld + col0 + oc] = v;
  }
}

// Store D frag (+bias) into f16 LDS tile.
__device__ __forceinline__ void stD_h(_Float16* base, int ld, int col0, int nt, v8f acc,
                                      const float* bias, int lane){
  int n  = lane & 15;
  int mb = (lane & 16) ? 8 : 0;
  int oc = nt * 16 + n;
  float bv = bias[oc];
  #pragma unroll
  for (int r = 0; r < 8; ++r)
    base[(mb + r) * ld + col0 + oc] = (_Float16)(acc[r] + bv);
}

template<int W>
__device__ __forceinline__ void ln_vec(float* v, const float* g, const float* b){
  float m = 0.f;
  #pragma unroll
  for (int c = 0; c < W; ++c) m += v[c];
  m *= (1.f / W);
  float var = 0.f;
  #pragma unroll
  for (int c = 0; c < W; ++c){ float d = v[c] - m; var += d * d; }
  var *= (1.f / W);
  float rs = rsqrtf(var + LN_EPS);
  #pragma unroll
  for (int c = 0; c < W; ++c) v[c] = (v[c] - m) * rs * g[c] + b[c];
}

// ---------------- parameter bundle ----------------
struct P {
  const float *x;
  const float *tp_w,*tp_b,*tp_g,*tp_bt;
  const float *mp_b,*mp_g,*mp_bt;
  const float *sa_wq,*sa_bq,*sa_bk,*sa_bv,*sa_bo,*sa_ng,*sa_nb;
  const float *ca_bv,*ca_bo,*ca_ng,*ca_nb;
  const float *ff_b1,*ff_b2,*ff_ng,*ff_nb;
  const float *pool_b,*pool_g,*pool_bt;
  const float *z_b1,*z_b2,*z_g,*z_bt;
  const float *y_b1,*y_b2,*y_g,*y_bt;
  const float *head_w,*head_b;
  const _Float16 *wpk;
  float *out;
};

// ---------------- weight pack kernel (f32 row-major -> f16 B fragments) ----------------
// B layout per ISA: lanes 0-15 (N = lane) hold K = kc*32 + {0..7,16..23}; lanes 16-31 hold K = kc*32 + {8..15,24..31}.
__global__ void pack_b(const float* __restrict__ src, _Float16* __restrict__ dst,
                       int K, int N, int NT, int total){
  int idx = blockIdx.x * 256 + threadIdx.x;
  if (idx >= total) return;
  int f  = idx >> 9, r = idx & 511;
  int ln = r >> 4,  j = r & 15;
  int kc = f / NT,  nt = f % NT;
  int k = kc * 32 + ((ln & 16) ? 8 : 0) + (j < 8 ? j : j + 8);
  int n = nt * 16 + (ln & 15);
  dst[idx] = (_Float16)((k < K && n < N) ? src[k * N + n] : 0.f);
}

// ---------------- main fused kernel ----------------
__global__ __launch_bounds__(128) void trm_main(P p){
  extern __shared__ __align__(16) char smem[];
  const int tid  = threadIdx.x;
  const int wave = tid >> 5;
  const int lane = tid & 31;
  const int tile = blockIdx.x * 4 + wave;
  const int s0   = tile * 16;

  char* wb = smem + (size_t)wave * WAVE_LDS;
  _Float16* tok    = (_Float16*)wb;              // 352 x 32 f16
  _Float16* kvh    = (_Float16*)(wb + 22528);    // 22528 halves
  float*    scr    = (float*)(wb + 67584);       // 16 x 128 f32
  float*    ctxo   = (float*)(wb + 75776);       // 16 x 32 f32
  float*    xstg   = (float*)(wb + 77824);       // 16 x 32 f32 TDM staging
  _Float16* Kb     = kvh;                        // [t*16+s][32]
  _Float16* Vb     = kvh + 11264;
  _Float16* zin    = kvh;                        // 16 x 192 f16 (post-attn reuse: [xp|y|z])
  float*    stateF = (float*)(wb + 30720);       // 16 x 128 f32 ([y|z] master)
  float*    tokF   = (float*)wb;                 // 16 x 64 f32 (recurrence reuse)

  // ===== Stage A: tokens = gelu(ln(magpie @ tp_w + tp_b)) , per-lane VALU (K=6) =====
  for (int it = 0; it < 11; ++it){
    int idx = it * 32 + lane;          // 0..351 == t*16+s
    int s = idx & 15, t = idx >> 4;
    const float* xr = p.x + (size_t)(s0 + s) * 332 + t * 6;
    __builtin_prefetch(xr, 0, 3);      // global_prefetch_b8
    float acc[32];
    #pragma unroll
    for (int n = 0; n < 32; ++n) acc[n] = p.tp_b[n];
    #pragma unroll
    for (int j = 0; j < 6; ++j){
      float xv = xr[j];
      #pragma unroll
      for (int n = 0; n < 32; ++n) acc[n] += xv * p.tp_w[j * 32 + n];
    }
    ln_vec<32>(acc, p.tp_g, p.tp_bt);
    _Float16* to = tok + idx * 32;
    #pragma unroll
    for (int n = 0; n < 32; ++n) to[n] = (_Float16)gelu_f(acc[n]);
  }
  lds_fence();

  // ===== Stage B: context = gelu(ln(m2v @ mp_w)); ctxo = (ctx@ca_wv+bv)@ca_wo+bo =====
  {
    v8f c0 = vzero(), c1 = vzero();
    const float* xbase = p.x + (size_t)s0 * 332 + 132;
    uint32_t stg_off = (uint32_t)(uintptr_t)xstg;   // flat LDS addr low 32 bits == LDS offset
    for (int kc = 0; kc < 7; ++kc){
#if __has_builtin(__builtin_amdgcn_tensor_load_to_lds)
      // TDM: 16 x 32 f32 tile; rem_cols drives hardware zero-fill past m2v's 200 cols
      tdm_load_2d_f32(xbase + kc * 32, stg_off, (uint32_t)(200 - kc * 32),
                      16u, 32u, 332u);
      tensor_wait();
#else
      for (int e = lane; e < 512; e += 32){
        int r = e >> 5, c = e & 31;
        int k = kc * 32 + c;
        xstg[r * 32 + c] = (k < 200) ? xbase[(size_t)r * 332 + k] : 0.f;
      }
      lds_fence();
#endif
      v16h a = ldA_f(xstg, 32, 0, lane);
      c0 = wmma_f16(a, ldB(p.wpk + OFF_MPW, kc * 2 + 0, lane), c0);
      c1 = wmma_f16(a, ldB(p.wpk + OFF_MPW, kc * 2 + 1, lane), c1);
      lds_fence();   // staging reads must land before next TDM overwrites it
    }
    stD_f(scr, 128, 0, 0, c0, p.mp_b, 32, false, lane);
    stD_f(scr, 128, 0, 1, c1, p.mp_b, 32, false, lane);
    lds_fence();
    if (lane < 16){
      float rv[32];
      float* rw = scr + lane * 128;
      #pragma unroll
      for (int c = 0; c < 32; ++c) rv[c] = rw[c];
      ln_vec<32>(rv, p.mp_g, p.mp_bt);
      #pragma unroll
      for (int c = 0; c < 32; ++c) rw[c] = gelu_f(rv[c]);
    }
    lds_fence();
    // val = ctx @ ca_wv + bv  (cross-attn softmax over 1 key == 1, so q/k dead)
    v16h a = ldA_f(scr, 128, 0, lane);
    v8f v0 = vzero(), v1 = vzero();
    v0 = wmma_f16(a, ldB(p.wpk + OFF_CAWV, 0, lane), v0);
    v1 = wmma_f16(a, ldB(p.wpk + OFF_CAWV, 1, lane), v1);
    stD_f(scr, 128, 32, 0, v0, p.ca_bv, 32, false, lane);
    stD_f(scr, 128, 32, 1, v1, p.ca_bv, 32, false, lane);
    lds_fence();
    v16h a2 = ldA_f(scr + 32, 128, 0, lane);
    v8f o0 = vzero(), o1 = vzero();
    o0 = wmma_f16(a2, ldB(p.wpk + OFF_CAWO, 0, lane), o0);
    o1 = wmma_f16(a2, ldB(p.wpk + OFF_CAWO, 1, lane), o1);
    stD_f(ctxo, 32, 0, 0, o0, p.ca_bo, 32, false, lane);
    stD_f(ctxo, 32, 0, 1, o1, p.ca_bo, 32, false, lane);
  }
  lds_fence();

  // ===== Stage C1: K/V projections via WMMA =====
  for (int t = 0; t < 22; ++t){
    v16h a = ldA_h(tok + t * 512, 32, 0, lane);
    v8f k0 = vzero(), k1 = vzero(), w0 = vzero(), w1 = vzero();
    k0 = wmma_f16(a, ldB(p.wpk + OFF_SAWK, 0, lane), k0);
    k1 = wmma_f16(a, ldB(p.wpk + OFF_SAWK, 1, lane), k1);
    w0 = wmma_f16(a, ldB(p.wpk + OFF_SAWV, 0, lane), w0);
    w1 = wmma_f16(a, ldB(p.wpk + OFF_SAWV, 1, lane), w1);
    stD_h(Kb + t * 512, 32, 0, 0, k0, p.sa_bk, lane);
    stD_h(Kb + t * 512, 32, 0, 1, k1, p.sa_bk, lane);
    stD_h(Vb + t * 512, 32, 0, 0, w0, p.sa_bv, lane);
    stD_h(Vb + t * 512, 32, 0, 1, w1, p.sa_bv, lane);
  }
  lds_fence();

  // ===== Stage C2: attention + out-proj + FF + cross-add, per token tile =====
  float psum[32];
  #pragma unroll
  for (int n = 0; n < 32; ++n) psum[n] = 0.f;
  const int as = lane >> 1, ah = lane & 1;   // 16 samples x 2 heads = 32 lanes

  for (int t = 0; t < 22; ++t){
    // q for (sample as, head ah), from current tok[t]
    float q[16];
    #pragma unroll
    for (int d = 0; d < 16; ++d) q[d] = p.sa_bq[ah * 16 + d];
    const _Float16* tr = tok + (t * 16 + as) * 32;
    for (int k = 0; k < 32; ++k){
      float tv = (float)tr[k];
      #pragma unroll
      for (int d = 0; d < 16; ++d) q[d] += tv * p.sa_wq[k * 32 + ah * 16 + d];
    }
    // scores + softmax (S=22, 1/sqrt(16)=0.25)
    float sc[22]; float mx = -3.0e30f;
    #pragma unroll
    for (int u = 0; u < 22; ++u){
      const _Float16* kr = Kb + (u * 16 + as) * 32 + ah * 16;
      float dth = 0.f;
      #pragma unroll
      for (int d = 0; d < 16; ++d) dth += q[d] * (float)kr[d];
      dth *= 0.25f;
      sc[u] = dth; mx = fmaxf(mx, dth);
    }
    float ssum = 0.f;
    #pragma unroll
    for (int u = 0; u < 22; ++u){ sc[u] = __expf(sc[u] - mx); ssum += sc[u]; }
    float inv = 1.f / ssum;
    float o[16];
    #pragma unroll
    for (int d = 0; d < 16; ++d) o[d] = 0.f;
    #pragma unroll
    for (int u = 0; u < 22; ++u){
      float w = sc[u] * inv;
      const _Float16* vr = Vb + (u * 16 + as) * 32 + ah * 16;
      #pragma unroll
      for (int d = 0; d < 16; ++d) o[d] += w * (float)vr[d];
    }
    #pragma unroll
    for (int d = 0; d < 16; ++d) scr[as * 128 + ah * 16 + d] = o[d];
    lds_fence();
    // o @ sa_wo + bo -> scr cols 32..63
    {
      v16h a = ldA_f(scr, 128, 0, lane);
      v8f w0 = vzero(), w1 = vzero();
      w0 = wmma_f16(a, ldB(p.wpk + OFF_SAWO, 0, lane), w0);
      w1 = wmma_f16(a, ldB(p.wpk + OFF_SAWO, 1, lane), w1);
      stD_f(scr, 128, 32, 0, w0, p.sa_bo, 32, false, lane);
      stD_f(scr, 128, 32, 1, w1, p.sa_bo, 32, false, lane);
    }
    lds_fence();
    // residual + LN(sa_norm) -> tok[t]
    if (lane < 16){
      float rv[32];
      _Float16* trw = tok + (t * 16 + lane) * 32;
      #pragma unroll
      for (int c = 0; c < 32; ++c) rv[c] = scr[lane * 128 + 32 + c] + (float)trw[c];
      ln_vec<32>(rv, p.sa_ng, p.sa_nb);
      #pragma unroll
      for (int c = 0; c < 32; ++c) trw[c] = (_Float16)rv[c];
    }
    lds_fence();
    // FF1: h = gelu(tok[t] @ w1 + b1) -> scr cols 64..127
    {
      v16h a = ldA_h(tok + t * 512, 32, 0, lane);
      #pragma unroll
      for (int nt = 0; nt < 4; ++nt){
        v8f h = vzero();
        h = wmma_f16(a, ldB(p.wpk + OFF_FF1, nt, lane), h);
        stD_f(scr, 128, 64, nt, h, p.ff_b1, 64, true, lane);
      }
    }
    lds_fence();
    // FF2: d = h @ w2 + b2 -> scr cols 32..63
    {
      v8f d0 = vzero(), d1 = vzero();
      for (int kc = 0; kc < 2; ++kc){
        v16h a = ldA_f(scr + 64, 128, kc, lane);
        d0 = wmma_f16(a, ldB(p.wpk + OFF_FF2, kc * 2 + 0, lane), d0);
        d1 = wmma_f16(a, ldB(p.wpk + OFF_FF2, kc * 2 + 1, lane), d1);
      }
      stD_f(scr, 128, 32, 0, d0, p.ff_b2, 32, false, lane);
      stD_f(scr, 128, 32, 1, d1, p.ff_b2, 32, false, lane);
    }
    lds_fence();
    // residual + LN(ff_norm); + ctxo; LN(ca_norm); write tok[t]; pool accumulate
    if (lane < 16){
      float rv[32];
      _Float16* trw = tok + (t * 16 + lane) * 32;
      #pragma unroll
      for (int c = 0; c < 32; ++c) rv[c] = scr[lane * 128 + 32 + c] + (float)trw[c];
      ln_vec<32>(rv, p.ff_ng, p.ff_nb);
      #pragma unroll
      for (int c = 0; c < 32; ++c) rv[c] += ctxo[lane * 32 + c];
      ln_vec<32>(rv, p.ca_ng, p.ca_nb);
      #pragma unroll
      for (int c = 0; c < 32; ++c){ trw[c] = (_Float16)rv[c]; psum[c] += rv[c]; }
    }
    lds_fence();
  }

  // ===== Stage D: pooling + xp = gelu(ln(pool @ pool_w + pool_b)); init state =====
  if (lane < 16){
    #pragma unroll
    for (int c = 0; c < 32; ++c) scr[lane * 128 + c] = psum[c] * (1.f / 22.f);
  }
  lds_fence();
  {
    v16h a = ldA_f(scr, 128, 0, lane);
    #pragma unroll
    for (int nt = 0; nt < 4; ++nt){
      v8f h = vzero();
      h = wmma_f16(a, ldB(p.wpk + OFF_POOL, nt, lane), h);
      stD_f(scr, 128, 64, nt, h, p.pool_b, 64, false, lane);
    }
  }
  lds_fence();
  if (lane < 16){
    float rv[64];
    float* rw = scr + lane * 128 + 64;
    #pragma unroll
    for (int c = 0; c < 64; ++c) rv[c] = rw[c];
    ln_vec<64>(rv, p.pool_g, p.pool_bt);
    _Float16* zr = zin + lane * 192;
    #pragma unroll
    for (int c = 0; c < 64; ++c) zr[c] = (_Float16)gelu_f(rv[c]);
    #pragma unroll
    for (int c = 64; c < 192; ++c) zr[c] = (_Float16)0.f;   // y = z = 0
    float* st = stateF + lane * 128;
    #pragma unroll
    for (int c = 0; c < 128; ++c) st[c] = 0.f;
    #pragma unroll
    for (int c = 112; c < 128; ++c) scr[lane * 128 + c] = 0.f; // zero K pad for GEMM2
  }
  lds_fence();

  // ===== Stage E: 16 recurrence steps =====
  for (int step = 0; step < 16; ++step){
    // ---- z: h = gelu(zin(192) @ z_w1 + b1) ; d = h @ z_w2 + b2 ; z += LN(d) ----
    {
      v8f acc[7];
      #pragma unroll
      for (int nt = 0; nt < 7; ++nt) acc[nt] = vzero();
      for (int kc = 0; kc < 6; ++kc){
        v16h a = ldA_h(zin, 192, kc, lane);
        #pragma unroll
        for (int nt = 0; nt < 7; ++nt)
          acc[nt] = wmma_f16(a, ldB(p.wpk + OFF_ZW1, kc * 7 + nt, lane), acc[nt]);
      }
      #pragma unroll
      for (int nt = 0; nt < 7; ++nt) stD_f(scr, 128, 0, nt, acc[nt], p.z_b1, 100, true, lane);
    }
    lds_fence();
    {
      v8f d[4];
      #pragma unroll
      for (int nt = 0; nt < 4; ++nt) d[nt] = vzero();
      for (int kc = 0; kc < 4; ++kc){
        v16h a = ldA_f(scr, 128, kc, lane);
        #pragma unroll
        for (int nt = 0; nt < 4; ++nt)
          d[nt] = wmma_f16(a, ldB(p.wpk + OFF_ZW2, kc * 4 + nt, lane), d[nt]);
      }
      #pragma unroll
      for (int nt = 0; nt < 4; ++nt) stD_f(tokF, 64, 0, nt, d[nt], p.z_b2, 64, false, lane);
    }
    lds_fence();
    if (lane < 16){
      float rv[64];
      float* rw = tokF + lane * 64;
      #pragma unroll
      for (int c = 0; c < 64; ++c) rv[c] = rw[c];
      ln_vec<64>(rv, p.z_g, p.z_bt);
      float* st = stateF + lane * 128;
      _Float16* zr = zin + lane * 192;
      #pragma unroll
      for (int c = 0; c < 64; ++c){
        float zn = st[64 + c] + rv[c];
        st[64 + c] = zn;
        zr[128 + c] = (_Float16)zn;
      }
    }
    lds_fence();
    // ---- y: h = gelu(yin(128) @ y_w1 + b1) ; d = h @ y_w2 + b2 ; y += LN(d) ----
    {
      v8f acc[7];
      #pragma unroll
      for (int nt = 0; nt < 7; ++nt) acc[nt] = vzero();
      for (int kc = 0; kc < 4; ++kc){
        v16h a = ldA_h(zin + 64, 192, kc, lane);
        #pragma unroll
        for (int nt = 0; nt < 7; ++nt)
          acc[nt] = wmma_f16(a, ldB(p.wpk + OFF_YW1, kc * 7 + nt, lane), acc[nt]);
      }
      #pragma unroll
      for (int nt = 0; nt < 7; ++nt) stD_f(scr, 128, 0, nt, acc[nt], p.y_b1, 100, true, lane);
    }
    lds_fence();
    {
      v8f d[4];
      #pragma unroll
      for (int nt = 0; nt < 4; ++nt) d[nt] = vzero();
      for (int kc = 0; kc < 4; ++kc){
        v16h a = ldA_f(scr, 128, kc, lane);
        #pragma unroll
        for (int nt = 0; nt < 4; ++nt)
          d[nt] = wmma_f16(a, ldB(p.wpk + OFF_YW2, kc * 4 + nt, lane), d[nt]);
      }
      #pragma unroll
      for (int nt = 0; nt < 4; ++nt) stD_f(tokF, 64, 0, nt, d[nt], p.y_b2, 64, false, lane);
    }
    lds_fence();
    if (lane < 16){
      float rv[64];
      float* rw = tokF + lane * 64;
      #pragma unroll
      for (int c = 0; c < 64; ++c) rv[c] = rw[c];
      ln_vec<64>(rv, p.y_g, p.y_bt);
      float* st = stateF + lane * 128;
      _Float16* zr = zin + lane * 192;
      #pragma unroll
      for (int c = 0; c < 64; ++c){
        float yn = st[c] + rv[c];
        st[c] = yn;
        zr[64 + c] = (_Float16)yn;
      }
    }
    lds_fence();
  }

  // ===== Stage F: head =====
  if (lane < 16){
    const float* st = stateF + lane * 128;
    float o = p.head_b[0];
    #pragma unroll
    for (int c = 0; c < 64; ++c) o += st[c] * p.head_w[c];
    p.out[s0 + lane] = o;
  }
}

// ---------------- host launcher ----------------
extern "C" void kernel_launch(void* const* d_in, const int* in_sizes, int n_in,
                              void* d_out, int out_size, void* d_ws, size_t ws_size,
                              hipStream_t stream){
  auto f = [&](int i){ return (const float*)d_in[i]; };
  _Float16* wpk = (_Float16*)d_ws;

  struct PKD { int idx, K, N, KC, NT; size_t off; };
  const PKD pkd[13] = {
    {5, 200,  32, 7, 2, OFF_MPW }, {11, 32, 32, 1, 2, OFF_SAWK},
    {13, 32,  32, 1, 2, OFF_SAWV}, {15, 32, 32, 1, 2, OFF_SAWO},
    {21, 32,  32, 1, 2, OFF_CAWV}, {23, 32, 32, 1, 2, OFF_CAWO},
    {27, 32,  64, 1, 4, OFF_FF1 }, {29, 64, 32, 2, 2, OFF_FF2 },
    {35, 32,  64, 1, 4, OFF_POOL}, {39, 192, 100, 6, 7, OFF_ZW1},
    {41, 100, 64, 4, 4, OFF_ZW2 }, {45, 128, 100, 4, 7, OFF_YW1},
    {47, 100, 64, 4, 4, OFF_YW2 }};
  for (int i = 0; i < 13; ++i){
    int total = pkd[i].KC * pkd[i].NT * 512;
    pack_b<<<(total + 255) / 256, 256, 0, stream>>>(
        f(pkd[i].idx), wpk + pkd[i].off, pkd[i].K, pkd[i].N, pkd[i].NT, total);
  }

  P prm;
  prm.x = f(0);
  prm.tp_w = f(1);  prm.tp_b = f(2);  prm.tp_g = f(3);  prm.tp_bt = f(4);
  prm.mp_b = f(6);  prm.mp_g = f(7);  prm.mp_bt = f(8);
  prm.sa_wq = f(9); prm.sa_bq = f(10); prm.sa_bk = f(12); prm.sa_bv = f(14);
  prm.sa_bo = f(16); prm.sa_ng = f(25); prm.sa_nb = f(26);
  prm.ca_bv = f(22); prm.ca_bo = f(24); prm.ca_ng = f(33); prm.ca_nb = f(34);
  prm.ff_b1 = f(28); prm.ff_b2 = f(30); prm.ff_ng = f(31); prm.ff_nb = f(32);
  prm.pool_b = f(36); prm.pool_g = f(37); prm.pool_bt = f(38);
  prm.z_b1 = f(40); prm.z_b2 = f(42); prm.z_g = f(43); prm.z_bt = f(44);
  prm.y_b1 = f(46); prm.y_b2 = f(48); prm.y_g = f(49); prm.y_bt = f(50);
  prm.head_w = f(51); prm.head_b = f(52);
  prm.wpk = wpk;
  prm.out = (float*)d_out;

  int Bsz = in_sizes[0] / 332;        // 65536
  int nblocks = Bsz / 64;             // 4 waves/block x 16 samples/wave
  trm_main<<<nblocks, 128, 4 * WAVE_LDS, stream>>>(prm);
}
